// ResGenerator_37443524886876
// MI455X (gfx1250) — matrix operand order repaired and checked by reading, not compiled, not run on hardware
//
#include <hip/hip_runtime.h>

typedef __bf16 bf16;
typedef __bf16 v16bf __attribute__((ext_vector_type(16)));
typedef __bf16 v8bf  __attribute__((ext_vector_type(8)));
typedef float  v8f   __attribute__((ext_vector_type(8)));

__device__ __forceinline__ v8bf zero8() {
  v8bf z;
#pragma unroll
  for (int i = 0; i < 8; ++i) z[i] = (bf16)0.0f;
  return z;
}
__device__ __forceinline__ v16bf join8(v8bf lo, v8bf hi) {
  return __builtin_shufflevector(lo, hi, 0, 1, 2, 3, 4, 5, 6, 7, 8, 9, 10, 11, 12, 13, 14, 15);
}

// ---------------------------------------------------------------------------
// Weight prep: f32 source tensor -> bf16 blocks in the B-fragment order this kernel uses (contiguous k).
// Block (kt, nt) holds a 32(K) x 16(N) tile as [lane][16 elems]:
//   lane L, elem e -> K = kt*32 + (L<16 ? 0 : 16) + e, N = nt*16 + (L%16)
// Zero-fill K/N padding.
// ---------------------------------------------------------------------------
__global__ __launch_bounds__(256) void prep_weights_kernel(
    const float* __restrict__ src, bf16* __restrict__ dst, int NT, int Cin_pad,
    int Cin_act, int Cout_act, int KW, int KHKW, int stride_n, int stride_ci,
    int total) {
  int tid = blockIdx.x * blockDim.x + threadIdx.x;
  if (tid >= total) return;
  int e = tid & 15;
  int lane = (tid >> 4) & 31;
  int blk = tid >> 9;
  int nt = blk % NT;
  int kt = blk / NT;
  int k = (kt << 5) + ((lane & 16) ? 16 : 0) + e;
  int n = (nt << 4) + (lane & 15);
  int tap = k / Cin_pad;
  int cin = k - tap * Cin_pad;
  float v = 0.0f;
  if (cin < Cin_act && n < Cout_act && tap < KHKW) {
    int kh = tap / KW, kw = tap - kh * KW;
    v = src[(size_t)n * stride_n + (size_t)cin * stride_ci + kh * KW + kw];
  }
  dst[tid] = (bf16)v;
}

// ---------------------------------------------------------------------------
// NCHW f32 -> NHWC bf16 with channel padding (zeros) and optional scalar add.
// ---------------------------------------------------------------------------
__global__ __launch_bounds__(256) void to_nhwc_kernel(
    const float* __restrict__ src, bf16* __restrict__ dst, int C_src,
    int C_take, int Cpad, int HW, float add, int total) {
  int tid = blockIdx.x * blockDim.x + threadIdx.x;
  if (tid >= total) return;
  int c = tid % Cpad;
  int p = tid / Cpad;
  int b = p / HW;
  int hw = p - b * HW;
  float v = 0.0f;
  if (c < C_take) v = src[((size_t)b * C_src + c) * HW + hw] + add;
  dst[tid] = (bf16)v;
}

// ---------------------------------------------------------------------------
// Generic implicit-GEMM conv with bf16 WMMA (16x16x32), f32 accumulate.
// mode: 0 = zero pad, 1 = reflect pad, 2 = ConvTranspose2d(k3,s2,p1,op1)
//
// Tile: 256 pixels (M) x 64 couts (N) per 256-thread block. Each of the 8
// waves owns 32 M-rows as two 16-row A fragments; the 32x64 B tile for the
// current K-step is staged into LDS (double-buffered, one barrier per step)
// and consumed by all waves via ds_read_b128. K is iterated as
// taps(outer, incremental kh/kw, gather address hoisted) x cin-chunks(inner,
// pointer bumps only) so the hot loop is loads + 8 WMMAs, no int div.
//
// Epilogue fuses bias, eval BN (g*rsqrt(1+eps), b), residual add, ReLU.
// N-padding channels (n >= Cout_act) are written as raw zeros.
// ---------------------------------------------------------------------------
__global__ __launch_bounds__(256) void conv_wmma_kernel(
    const bf16* __restrict__ in, const bf16* __restrict__ wfrag,
    bf16* __restrict__ out, const float* __restrict__ bias,
    const float* __restrict__ bn_g, const float* __restrict__ bn_b,
    const bf16* __restrict__ res, int Bn, int Hin, int Win, int Hout, int Wout,
    int Cin_pad, int Cout_pad, int Cout_act, int KH, int KW, int stride,
    int pad, int mode, int relu_flag, int out_cs, int out_off, int res_cs) {
  __shared__ bf16 ldsB[2][2048];  // 2 x 4KB double-buffered B tile (32x64)

  const int tid = threadIdx.x;
  const int lane = tid & 31;
  const int wave = tid >> 5;
  const int M = Bn * Hout * Wout;
  const int mBase = blockIdx.x * 256 + wave * 32;
  const int nBase = blockIdx.y * 64;
  const int NT = Cout_pad >> 4;
  const int KHKW = KH * KW;
  const int koffA = (lane & 16) ? 16 : 0;

  // Decode the two A-fragment rows owned by this lane.
  int ox[2], oy[2], ob[2];
#pragma unroll
  for (int f = 0; f < 2; ++f) {
    int mA = mBase + f * 16 + (lane & 15);
    if (mA >= M) mA = M - 1;
    ox[f] = mA % Wout;
    oy[f] = (mA / Wout) % Hout;
    ob[f] = mA / (Wout * Hout);
  }

  v8f acc[2][4];
#pragma unroll
  for (int f = 0; f < 2; ++f)
#pragma unroll
    for (int t = 0; t < 4; ++t)
#pragma unroll
      for (int r = 0; r < 8; ++r) acc[f][t][r] = 0.0f;

  const bf16* wPtr = wfrag + (size_t)(nBase >> 4) * 512;  // (kt=0, nt0) block
  const size_t wStep = (size_t)NT * 512;                  // advance one kt
  int pb = 0;
  int kh = 0, kw = 0;

  for (int tap = 0; tap < KHKW; ++tap) {
    // Per-tap gather setup for both A fragments (hoisted out of cin loop).
    const bf16* ap[2];
    bool av[2];
#pragma unroll
    for (int f = 0; f < 2; ++f) {
      int iy, ix;
      bool valid = true;
      if (mode == 2) {  // transposed conv: oy = 2*iy + kh - 1
        int ty = oy[f] + 1 - kh, tx = ox[f] + 1 - kw;
        valid = (ty >= 0) && (tx >= 0) && !(ty & 1) && !(tx & 1);
        iy = ty >> 1;
        ix = tx >> 1;
        valid = valid && (iy < Hin) && (ix < Win);
      } else {
        iy = oy[f] * stride + kh - pad;
        ix = ox[f] * stride + kw - pad;
        if (mode == 1) {  // reflect pad 1
          iy = iy < 0 ? -iy : (iy >= Hin ? 2 * Hin - 2 - iy : iy);
          ix = ix < 0 ? -ix : (ix >= Win ? 2 * Win - 2 - ix : ix);
        } else {
          valid = (iy >= 0) && (iy < Hin) && (ix >= 0) && (ix < Win);
        }
      }
      av[f] = valid;
      ap[f] = in + ((size_t)((ob[f] * Hin + iy) * Win + ix)) * Cin_pad + koffA;
    }

    for (int c = 0; c < Cin_pad; c += 32) {
      // Stage this K-step's 32x64 B tile into LDS (one b128 per thread),
      // and prefetch the next tile while WMMAs run.
      *(v8bf*)&ldsB[pb][tid * 8] = *(const v8bf*)(wPtr + tid * 8);
      __builtin_prefetch(wPtr + wStep + tid * 8, 0, 0);
      __syncthreads();

      // A fragments: two 16B global loads each (zero if gather invalid).
      v16bf afrag[2];
#pragma unroll
      for (int f = 0; f < 2; ++f) {
        v8bf a0 = zero8(), a1 = zero8();
        if (av[f]) {
          const bf16* p = ap[f] + c;
          a0 = *(const v8bf*)(p);
          a1 = *(const v8bf*)(p + 8);
        }
        afrag[f] = join8(a0, a1);
      }

#pragma unroll
      for (int t = 0; t < 4; ++t) {
        const bf16* wp = &ldsB[pb][t * 512 + lane * 16];
        v16bf bfrag = join8(*(const v8bf*)wp, *(const v8bf*)(wp + 8));
        acc[0][t] = __builtin_amdgcn_wmma_f32_16x16x32_bf16(
            false, afrag[0], false, bfrag, (short)0, acc[0][t], false, false);
        acc[1][t] = __builtin_amdgcn_wmma_f32_16x16x32_bf16(
            false, afrag[1], false, bfrag, (short)0, acc[1][t], false, false);
      }
      pb ^= 1;
      wPtr += wStep;
    }
    if (++kw == KW) { kw = 0; ++kh; }
  }

  // Epilogue. C/D layout: VGPR r, lane L -> M = r + (L<16?0:8), N = L%16.
  const float invs = rsqrtf(1.0f + 1e-5f);
#pragma unroll
  for (int f = 0; f < 2; ++f) {
    const int mRow = mBase + f * 16 + ((lane & 16) ? 8 : 0);
#pragma unroll
    for (int t = 0; t < 4; ++t) {
      const int n = nBase + t * 16 + (lane & 15);
#pragma unroll
      for (int r = 0; r < 8; ++r) {
        const int m = mRow + r;
        if (m >= M) continue;
        float v = acc[f][t][r];
        if (n < Cout_act) {
          if (bias) v += bias[n];
          if (bn_g) v = v * (bn_g[n] * invs) + bn_b[n];
          if (res) v += (float)res[(size_t)m * res_cs + n];
          if (relu_flag) v = v > 0.0f ? v : 0.0f;
        } else {
          v = 0.0f;  // real zeros in channel padding (feeds next layer's K pad)
        }
        out[(size_t)m * out_cs + out_off + n] = (bf16)v;
      }
    }
  }
}

// ---------------------------------------------------------------------------
// FlowNet correlation: corr[b,d,y,x] = (1/64) sum_c f1[b,c,y,x]*f2 shifted.
// d < 441: dy=(d/21)*2-20, dx=(d%21)*2-20 (zero outside). d in [441,448): 0.
// Output NHWC bf16, channel stride 448.
// ---------------------------------------------------------------------------
__global__ __launch_bounds__(256) void corr_kernel(
    const bf16* __restrict__ f1, const bf16* __restrict__ f2,
    bf16* __restrict__ out, int Hh, int Ww, int total) {
  int tid = blockIdx.x * blockDim.x + threadIdx.x;
  if (tid >= total) return;
  const int Dpad = 448, C = 64;
  int d = tid % Dpad;
  int p = tid / Dpad;
  int x = p % Ww;
  int y = (p / Ww) % Hh;
  int b = p / (Ww * Hh);
  float s = 0.0f;
  if (d < 441) {
    int dy = (d / 21) * 2 - 20, dx = (d % 21) * 2 - 20;
    int y2 = y + dy, x2 = x + dx;
    if (y2 >= 0 && y2 < Hh && x2 >= 0 && x2 < Ww) {
      const bf16* p1 = f1 + (size_t)p * C;
      const bf16* p2 = f2 + ((size_t)(b * Hh + y2) * Ww + x2) * C;
      for (int c = 0; c < C; c += 8) {
        v8bf u = *(const v8bf*)(p1 + c);
        v8bf w = *(const v8bf*)(p2 + c);
#pragma unroll
        for (int j = 0; j < 8; ++j) s += (float)u[j] * (float)w[j];
      }
    }
    s *= (1.0f / 64.0f);
  }
  out[tid] = (bf16)s;
}

// ---------------------------------------------------------------------------
// Align-corners bilinear x2: NHWC bf16 (Cpad) -> NCHW f32 output.
// ---------------------------------------------------------------------------
__global__ __launch_bounds__(256) void upsample_kernel(
    const bf16* __restrict__ gen, float* __restrict__ out, int C_act, int Cpad,
    int Hin, int Win, int Hout, int Wout, int total) {
  int tid = blockIdx.x * blockDim.x + threadIdx.x;
  if (tid >= total) return;
  int x = tid % Wout;
  int y = (tid / Wout) % Hout;
  int c = (tid / (Wout * Hout)) % C_act;
  int b = tid / (Wout * Hout * C_act);
  float fy = (float)y * (float)(Hin - 1) / (float)(Hout - 1);
  float fx = (float)x * (float)(Win - 1) / (float)(Wout - 1);
  int y0 = (int)fy, x0 = (int)fx;
  int y1 = y0 + 1 < Hin ? y0 + 1 : Hin - 1;
  int x1 = x0 + 1 < Win ? x0 + 1 : Win - 1;
  float wy = fy - (float)y0, wx = fx - (float)x0;
  float g00 = (float)gen[((size_t)(b * Hin + y0) * Win + x0) * Cpad + c];
  float g01 = (float)gen[((size_t)(b * Hin + y0) * Win + x1) * Cpad + c];
  float g10 = (float)gen[((size_t)(b * Hin + y1) * Win + x0) * Cpad + c];
  float g11 = (float)gen[((size_t)(b * Hin + y1) * Win + x1) * Cpad + c];
  float v = g00 * (1.0f - wy) * (1.0f - wx) + g01 * (1.0f - wy) * wx +
            g10 * wy * (1.0f - wx) + g11 * wy * wx;
  out[tid] = v;
}

// ---------------------------------------------------------------------------
// Host launcher
// ---------------------------------------------------------------------------
extern "C" void kernel_launch(void* const* d_in, const int* in_sizes, int n_in,
                              void* d_out, int out_size, void* d_ws,
                              size_t ws_size, hipStream_t stream) {
  (void)in_sizes; (void)n_in; (void)out_size; (void)ws_size;
  const float* gt = (const float*)d_in[0];
  const float* fr = (const float*)d_in[1];
  auto F = [&](int i) { return (const float*)d_in[i]; };

  char* base = (char*)d_ws;
  size_t off = 0;
  auto alloc = [&](size_t bytes) -> void* {
    size_t cur = (off + 255) & ~(size_t)255;
    off = cur + bytes;
    return base + cur;
  };

  auto prep = [&](const float* w, int Cin_act, int Cin_pad, int Cout_act,
                  int Cout_pad, int KH, int KW, int transpose) -> bf16* {
    int K = KH * KW * Cin_pad, KT = K / 32, NT = Cout_pad / 16;
    int total = KT * NT * 512;
    bf16* dst = (bf16*)alloc((size_t)total * sizeof(bf16));
    int sn, sc;
    if (transpose) { sn = KH * KW; sc = Cout_act * KH * KW; }        // [ci][co][kh][kw]
    else           { sn = Cin_act * KH * KW; sc = KH * KW; }          // [co][ci][kh][kw]
    prep_weights_kernel<<<(total + 255) / 256, 256, 0, stream>>>(
        w, dst, NT, Cin_pad, Cin_act, Cout_act, KW, KH * KW, sn, sc, total);
    return dst;
  };

  auto conv = [&](const bf16* in, const bf16* wf, bf16* out, const float* bias,
                  const float* g, const float* bb, const bf16* res, int Bn,
                  int Hin, int Win, int Hout, int Wout, int Cin_pad,
                  int Cout_pad, int Cout_act, int KH, int stride, int pad,
                  int mode, int relu, int out_cs, int out_off, int res_cs) {
    int M = Bn * Hout * Wout;
    dim3 grid((M + 255) / 256, Cout_pad / 64);
    conv_wmma_kernel<<<grid, 256, 0, stream>>>(
        in, wf, out, bias, g, bb, res, Bn, Hin, Win, Hout, Wout, Cin_pad,
        Cout_pad, Cout_act, KH, KH, stride, pad, mode, relu, out_cs, out_off,
        res_cs);
  };

  const int B2 = 2, H = 256, W = 256, H2 = 128, W2 = 128, H4 = 64, W4 = 64,
            H8 = 32, W8 = 32;

  // ---- input conversion (NCHW f32 -> NHWC bf16, Cpad=32) ----
  bf16* xF = (bf16*)alloc((size_t)B2 * H * W * 32 * 2);
  bf16* xL = (bf16*)alloc((size_t)B2 * H * W * 32 * 2);
  {
    int tot = B2 * H * W * 32;
    to_nhwc_kernel<<<(tot + 255) / 256, 256, 0, stream>>>(fr, xF, 3, 3, 32,
                                                          H * W, 0.0f, tot);
    to_nhwc_kernel<<<(tot + 255) / 256, 256, 0, stream>>>(gt, xL, 3, 1, 32,
                                                          H * W, -50.0f, tot);
  }

  // ---- weight prep (fragment-swizzled bf16) ----
  bf16* w_c11f = prep(F(2), 3, 32, 64, 64, 3, 3, 0);
  bf16* w_c12f = prep(F(6), 64, 64, 64, 64, 3, 3, 0);
  bf16* w_c11l = prep(F(10), 1, 32, 64, 64, 3, 3, 0);
  bf16* w_c12l = prep(F(14), 64, 64, 64, 64, 3, 3, 0);
  bf16* w_a1 = prep(F(18), 64, 64, 64, 64, 1, 1, 0);
  bf16* w_a2 = prep(F(22), 441, 448, 32, 64, 1, 1, 0);
  bf16* w_c0 = prep(F(26), 96, 96, 64, 64, 3, 3, 0);
  bf16* w_d0 = prep(F(29), 64, 64, 128, 128, 3, 3, 0);
  bf16* w_d1 = prep(F(32), 128, 128, 256, 256, 3, 3, 0);
  bf16* w_r[13][2];
  for (int i = 0; i < 13; ++i) {
    w_r[i][0] = prep(F(35 + 6 * i), 256, 256, 256, 256, 3, 3, 0);
    w_r[i][1] = prep(F(38 + 6 * i), 256, 256, 256, 256, 3, 3, 0);
  }
  bf16* w_u0 = prep(F(113), 256, 256, 128, 128, 3, 3, 1);
  bf16* w_u1 = prep(F(116), 128, 128, 64, 64, 3, 3, 1);
  bf16* w_cf = prep(F(119), 64, 64, 313, 320, 3, 3, 0);

  // ---- activation buffers ----
  bf16* hF1 = (bf16*)alloc((size_t)B2 * H * W * 64 * 2);
  bf16* hF2 = (bf16*)alloc((size_t)B2 * H2 * W2 * 64 * 2);
  bf16* hL1 = (bf16*)alloc((size_t)B2 * H * W * 64 * 2);
  bf16* hL2 = (bf16*)alloc((size_t)B2 * H2 * W2 * 64 * 2);
  bf16* corr = (bf16*)alloc((size_t)B2 * H2 * W2 * 448 * 2);
  bf16* cat = (bf16*)alloc((size_t)B2 * H2 * W2 * 96 * 2);
  bf16* g0 = (bf16*)alloc((size_t)B2 * H2 * W2 * 64 * 2);
  bf16* g1 = (bf16*)alloc((size_t)B2 * H4 * W4 * 128 * 2);
  bf16* hA = (bf16*)alloc((size_t)B2 * H8 * W8 * 256 * 2);
  bf16* hB = (bf16*)alloc((size_t)B2 * H8 * W8 * 256 * 2);
  bf16* tmp = (bf16*)alloc((size_t)B2 * H8 * W8 * 256 * 2);
  bf16* u0b = (bf16*)alloc((size_t)B2 * H4 * W4 * 128 * 2);
  bf16* u1b = (bf16*)alloc((size_t)B2 * H2 * W2 * 64 * 2);
  bf16* genb = (bf16*)alloc((size_t)B2 * H2 * W2 * 320 * 2);

  // ---- encoders ----
  conv(xF, w_c11f, hF1, F(3), F(4), F(5), nullptr, B2, H, W, H, W, 32, 64, 64,
       3, 1, 1, 0, 1, 64, 0, 0);
  conv(hF1, w_c12f, hF2, F(7), F(8), F(9), nullptr, B2, H, W, H2, W2, 64, 64,
       64, 3, 2, 1, 0, 1, 64, 0, 0);
  conv(xL, w_c11l, hL1, F(11), F(12), F(13), nullptr, B2, H, W, H, W, 32, 64,
       64, 3, 1, 1, 0, 1, 64, 0, 0);
  conv(hL1, w_c12l, hL2, F(15), F(16), F(17), nullptr, B2, H, W, H2, W2, 64,
       64, 64, 3, 2, 1, 0, 1, 64, 0, 0);

  // ---- correlation (f1 = frame branch, f2 = L branch) ----
  {
    int tot = B2 * H2 * W2 * 448;
    corr_kernel<<<(tot + 255) / 256, 256, 0, stream>>>(hF2, hL2, corr, H2, W2,
                                                       tot);
  }

  // ---- fuse: cat = [a2(corr) 32ch | a1(hL2) 64ch], stride 96 ----
  conv(corr, w_a2, cat, F(23), F(24), F(25), nullptr, B2, H2, W2, H2, W2, 448,
       64, 32, 1, 1, 0, 0, 1, 96, 0, 0);
  conv(hL2, w_a1, cat, F(19), F(20), F(21), nullptr, B2, H2, W2, H2, W2, 64,
       64, 64, 1, 1, 0, 0, 1, 96, 32, 0);

  // ---- ResnetGenerator ----
  conv(cat, w_c0, g0, nullptr, F(27), F(28), nullptr, B2, H2, W2, H2, W2, 96,
       64, 64, 3, 1, 1, 1, 1, 64, 0, 0);
  conv(g0, w_d0, g1, nullptr, F(30), F(31), nullptr, B2, H2, W2, H4, W4, 64,
       128, 128, 3, 2, 1, 0, 1, 128, 0, 0);
  conv(g1, w_d1, hA, nullptr, F(33), F(34), nullptr, B2, H4, W4, H8, W8, 128,
       256, 256, 3, 2, 1, 0, 1, 256, 0, 0);

  for (int i = 0; i < 13; ++i) {
    conv(hA, w_r[i][0], tmp, nullptr, F(36 + 6 * i), F(37 + 6 * i), nullptr,
         B2, H8, W8, H8, W8, 256, 256, 256, 3, 1, 1, 1, 1, 256, 0, 0);
    conv(tmp, w_r[i][1], hB, nullptr, F(39 + 6 * i), F(40 + 6 * i), hA, B2, H8,
         W8, H8, W8, 256, 256, 256, 3, 1, 1, 1, 0, 256, 0, 256);
    bf16* t = hA; hA = hB; hB = t;
  }

  conv(hA, w_u0, u0b, nullptr, F(114), F(115), nullptr, B2, H8, W8, H4, W4,
       256, 128, 128, 3, 2, 1, 2, 1, 128, 0, 0);
  conv(u0b, w_u1, u1b, nullptr, F(117), F(118), nullptr, B2, H4, W4, H2, W2,
       128, 64, 64, 3, 2, 1, 2, 1, 64, 0, 0);
  conv(u1b, w_cf, genb, F(120), nullptr, nullptr, nullptr, B2, H2, W2, H2, W2,
       64, 320, 313, 3, 1, 1, 1, 1, 320, 0, 0);

  // ---- bilinear x2 (align corners) -> f32 NCHW output ----
  {
    int tot = B2 * 313 * H * W;
    upsample_kernel<<<(tot + 255) / 256, 256, 0, stream>>>(
        genb, (float*)d_out, 313, 320, H2, W2, H, W, tot);
  }
}
